// OONativeModel_59966333386874
// MI455X (gfx1250) — matrix operand order repaired and checked by reading, not compiled, not run on hardware
//
#include <hip/hip_runtime.h>
#include <hip/hip_bf16.h>

// ---------------- model constants (from reference) ----------------
#define V_     16384
#define DM_    512
#define NLAYER 8
#define DS_    16
#define DC_    4
#define DI_    1024          // DM*2
#define DTR_   32
#define B_     2
#define L_     1024
#define BL_    (B_ * L_)     // 2048

typedef __bf16 bf16_t;
typedef __attribute__((ext_vector_type(16))) __bf16 v16bf;
typedef __attribute__((ext_vector_type(8)))  __bf16 v8bf;
typedef __attribute__((ext_vector_type(8)))  float  v8f;

// ---------------- helpers ----------------
__device__ __forceinline__ float sigm(float x) { return 1.f / (1.f + expf(-x)); }
__device__ __forceinline__ float softplus_f(float x) {
    return (x > 20.f) ? x : log1pf(expf(x));
}

// Async copy of 16 bytes global -> LDS (per lane). Tracked by ASYNCcnt.
__device__ __forceinline__ void async_b128(uint32_t lds_off, uint64_t gaddr) {
    asm volatile("global_load_async_to_lds_b128 %0, %1, off"
                 :: "v"(lds_off), "v"(gaddr) : "memory");
}
__device__ __forceinline__ void wait_async0() {
    asm volatile("s_wait_asynccnt 0x0" ::: "memory");
}

// A fragment: 16x32 bf16 (MxK). Lane l<16 holds row M=l, K=0..7 (elems 0..7)
// and K=16..23 (elems 8..15); lane l+16 holds K=8..15 and 24..31.
__device__ __forceinline__ v16bf load_frag_a(const bf16_t* A, int lda, int row0,
                                             int lane) {
    int r  = lane & 15;
    int hi = lane >> 4;
    const bf16_t* p = A + (size_t)(row0 + r) * lda + hi * 8;
    v8bf lo = *(const v8bf*)(p);
    v8bf hh = *(const v8bf*)(p + 16);
    v16bf f;
#pragma unroll
    for (int i = 0; i < 8; i++) { f[i] = lo[i]; f[i + 8] = hh[i]; }
    return f;
}

// B fragment: 32x16 bf16 (KxN), sourced from weight-matrix rows W[N][K].
// Lane n<16 holds column n, K=0..7 and 16..23; lane n+16 holds K=8..15 and 24..31.
__device__ __forceinline__ v16bf load_frag_b(const bf16_t* W, int ldb, int col0,
                                             int lane) {
    int n  = lane & 15;
    int hi = lane >> 4;
    const bf16_t* p = W + (size_t)(col0 + n) * ldb + hi * 8;
    v8bf lo = *(const v8bf*)(p);
    v8bf hh = *(const v8bf*)(p + 16);
    v16bf f;
#pragma unroll
    for (int i = 0; i < 8; i++) { f[i] = lo[i]; f[i + 8] = hh[i]; }
    return f;
}

// ------------- LDS-staged bf16 WMMA GEMM: C[M,N] = A[M,K] * W[N,K]^T --------
// 128 threads = 4 waves; block computes a 64x64 tile, K-step 32.
// A/B tiles staged into LDS via GLOBAL_LOAD_ASYNC_TO_LDS_B128 (double buffer,
// ASYNCcnt + workgroup barrier), WMMA operands fed from LDS (ds_load_b128).
template <bool BIAS, bool RES, int ACT, bool WBF>
__global__ __launch_bounds__(128) void k_gemm(const bf16_t* __restrict__ A,
                                              const bf16_t* __restrict__ W,
                                              float* __restrict__ C,
                                              bf16_t* __restrict__ Cb,
                                              const float* __restrict__ bias,
                                              const float* __restrict__ res,
                                              int M, int N, int K, int lda) {
    __shared__ __align__(16) bf16_t shA[2][64 * 32];  // 64 rows x 32 k (packed)
    __shared__ __align__(16) bf16_t shB[2][64 * 32];  // 64 cols x 32 k (packed)

    int tid  = threadIdx.x;
    int lane = tid & 31;
    int wave = tid >> 5;
    int mrow = blockIdx.y * 64;       // block row base
    int n0   = blockIdx.x * 64;       // block col base
    int m0   = mrow + wave * 16;      // wave row base

    const char* Ab = (const char*)A;
    const char* Wb = (const char*)W;

    // stage one 64x32 A tile + 64x32 B tile into LDS buffer `buf` (async)
    auto stage = [&](int buf, int k0) {
#pragma unroll
        for (int i = 0; i < 2; i++) {
            int c    = tid + i * 128;      // chunk 0..255, 16B each
            int row  = c >> 2;             // 0..63
            int bofs = (c & 3) * 16;       // byte offset within 64B row
            uint64_t ga = (uint64_t)(uintptr_t)(Ab +
                ((size_t)(mrow + row) * lda + k0) * 2 + bofs);
            uint32_t la = (uint32_t)(uintptr_t)(void*)(&shA[buf][row * 32]) + bofs;
            async_b128(la, ga);
            uint64_t gb = (uint64_t)(uintptr_t)(Wb +
                ((size_t)(n0 + row) * K + k0) * 2 + bofs);
            uint32_t lb = (uint32_t)(uintptr_t)(void*)(&shB[buf][row * 32]) + bofs;
            async_b128(lb, gb);
        }
    };

    v8f acc[4] = {v8f{}, v8f{}, v8f{}, v8f{}};

    int nk = K >> 5;
    stage(0, 0);
    for (int kk = 0; kk < nk; kk++) {
        wait_async0();        // my wave's async tile chunks have landed in LDS
        __syncthreads();      // publish tile; also fences reuse of prev buffer
        if (kk + 1 < nk) stage((kk + 1) & 1, (kk + 1) * 32);

        const bf16_t* sA = shA[kk & 1];
        const bf16_t* sB = shB[kk & 1];
        v16bf a  = load_frag_a(sA, 32, wave * 16, lane);
        v16bf b0 = load_frag_b(sB, 32,  0, lane);
        v16bf b1 = load_frag_b(sB, 32, 16, lane);
        v16bf b2 = load_frag_b(sB, 32, 32, lane);
        v16bf b3 = load_frag_b(sB, 32, 48, lane);
        acc[0] = __builtin_amdgcn_wmma_f32_16x16x32_bf16(false, a, false, b0,
                                                         (short)0, acc[0], false, false);
        acc[1] = __builtin_amdgcn_wmma_f32_16x16x32_bf16(false, a, false, b1,
                                                         (short)0, acc[1], false, false);
        acc[2] = __builtin_amdgcn_wmma_f32_16x16x32_bf16(false, a, false, b2,
                                                         (short)0, acc[2], false, false);
        acc[3] = __builtin_amdgcn_wmma_f32_16x16x32_bf16(false, a, false, b3,
                                                         (short)0, acc[3], false, false);
    }

    // D layout: VGPR r -> M = r (lanes 0-15) / M = r+8 (lanes 16-31); N = lane&15
    int rbase = m0 + (lane >> 4) * 8;
#pragma unroll
    for (int j = 0; j < 4; j++) {
        int col = n0 + j * 16 + (lane & 15);
#pragma unroll
        for (int r = 0; r < 8; r++) {
            int   row = rbase + r;
            float v   = acc[j][r];
            if (BIAS) v += bias[col];
            if (ACT == 1) v = softplus_f(v);
            if (RES) v += res[(size_t)row * N + col];
            C[(size_t)row * N + col] = v;
            if (WBF) Cb[(size_t)row * N + col] = (bf16_t)v;
        }
    }
}

// ---------------- fp32 -> bf16 conversion ----------------
__global__ void k_f32_to_bf16(const float* __restrict__ src,
                              bf16_t* __restrict__ dst, int n) {
    int i = blockIdx.x * blockDim.x + threadIdx.x;
    if (i < n) dst[i] = (bf16_t)src[i];
}

// ---------------- embedding + positional ----------------
__global__ void k_embed(const int* __restrict__ ids,
                        const float* __restrict__ emb,
                        const float* __restrict__ pos,
                        float* __restrict__ h) {
    int idx = blockIdx.x * blockDim.x + threadIdx.x;  // over B*L*DM
    if (idx >= BL_ * DM_) return;
    int d  = idx % DM_;
    int bl = idx / DM_;
    int l  = bl % L_;
    int tk = ids[bl];
    h[idx] = emb[(size_t)tk * DM_ + d] + pos[(size_t)l * DM_ + d];
}

// ---------------- LayerNorm (row per block), bf16 output ----------------
__global__ __launch_bounds__(128) void k_layernorm(const float* __restrict__ X,
                                                   const float* __restrict__ w,
                                                   const float* __restrict__ b,
                                                   bf16_t* __restrict__ out,
                                                   int D) {
    int row = blockIdx.x;
    const float* x = X + (size_t)row * D;
    float s1 = 0.f, s2 = 0.f;
    for (int c = threadIdx.x; c < D; c += 128) {
        float v = x[c];
        s1 += v;
        s2 += v * v;
    }
    __shared__ float r1[128], r2[128];
    r1[threadIdx.x] = s1;
    r2[threadIdx.x] = s2;
    __syncthreads();
    for (int off = 64; off > 0; off >>= 1) {
        if (threadIdx.x < off) {
            r1[threadIdx.x] += r1[threadIdx.x + off];
            r2[threadIdx.x] += r2[threadIdx.x + off];
        }
        __syncthreads();
    }
    float m   = r1[0] / (float)D;
    float var = r2[0] / (float)D - m * m;
    float rs  = rsqrtf(var + 1e-5f);
    for (int c = threadIdx.x; c < D; c += 128) {
        out[(size_t)row * D + c] = (bf16_t)((x[c] - m) * rs * w[c] + b[c]);
    }
}

// ---------------- depthwise causal conv (DC=4) + bias + SiLU ----------------
__global__ void k_conv_silu(const float* __restrict__ xz,
                            const float* __restrict__ cw,
                            const float* __restrict__ cb,
                            float* __restrict__ xf,
                            bf16_t* __restrict__ xfb) {
    int idx = blockIdx.x * blockDim.x + threadIdx.x;  // over B*L*DI
    if (idx >= BL_ * DI_) return;
    int d  = idx % DI_;
    int bl = idx / DI_;
    int l  = bl % L_;
    int bb = bl / L_;
    float acc = cb[d];
#pragma unroll
    for (int j = 0; j < DC_; j++) {
        int t = l - (DC_ - 1) + j;
        if (t >= 0)
            acc += cw[d * DC_ + j] * xz[(size_t)(bb * L_ + t) * (2 * DI_) + d];
    }
    float s  = acc * sigm(acc);
    xf[idx]  = s;
    xfb[idx] = (bf16_t)s;
}

// ---------------- selective scan (reference's cum_dA formulation) -----------
// one thread per (b, channel); sequential over L; DS=16 states in registers.
__global__ __launch_bounds__(256) void k_scan(const float* __restrict__ dt,
                                              const float* __restrict__ xf,
                                              const float* __restrict__ dtbc,
                                              const float* __restrict__ xz,
                                              const float* __restrict__ a_log,
                                              const float* __restrict__ Dp,
                                              bf16_t* __restrict__ yb) {
    int g = blockIdx.x * 256 + threadIdx.x;  // over B*DI
    if (g >= B_ * DI_) return;
    int d  = g % DI_;
    int bb = g / DI_;

    float As[DS_];
#pragma unroll
    for (int s = 0; s < DS_; s++) As[s] = -expf(a_log[d * DS_ + s]);
    float Dd = Dp[d];

    float ldA[DS_], S[DS_];
#pragma unroll
    for (int s = 0; s < DS_; s++) { ldA[s] = 0.f; S[s] = 0.f; }

    const float LOG1EM8 = -18.420680743952367f;  // log(1e-8)
    for (int t = 0; t < L_; t++) {
        size_t rm  = (size_t)(bb * L_ + t);
        float  dtv = dt[rm * DI_ + d];
        float  xv  = xf[rm * DI_ + d];
        const float* bc = dtbc + rm * 64;  // [dt_lo(32) | B(16) | C(16)]
        float y = 0.f;
#pragma unroll
        for (int s = 0; s < DS_; s++) {
            ldA[s] += fmaxf(dtv * As[s], LOG1EM8);       // log(max(dA,1e-8))
            float cum = expf(ldA[s]);                    // cum_dA
            float Bu  = dtv * bc[32 + s] * xv;
            S[s] += Bu / fmaxf(cum, 1e-8f);
            y += cum * S[s] * bc[48 + s];
        }
        y += xv * Dd;
        float z = xz[rm * (2 * DI_) + DI_ + d];
        y *= z * sigm(z);
        yb[rm * DI_ + d] = (bf16_t)y;
    }
}

// ---------------- host-side orchestration ----------------
static inline size_t align_up(size_t x) { return (x + 255) & ~(size_t)255; }

extern "C" void kernel_launch(void* const* d_in, const int* in_sizes, int n_in,
                              void* d_out, int out_size, void* d_ws, size_t ws_size,
                              hipStream_t stream) {
    (void)in_sizes; (void)n_in; (void)out_size; (void)ws_size;

    const int*   ids     = (const int*)d_in[0];
    const float* emb     = (const float*)d_in[1];
    const float* pos     = (const float*)d_in[2];
    const float* norm_w  = (const float*)d_in[3];
    const float* norm_b  = (const float*)d_in[4];
    const float* ipw     = (const float*)d_in[5];
    const float* conv_w  = (const float*)d_in[6];
    const float* conv_b  = (const float*)d_in[7];
    const float* xpw     = (const float*)d_in[8];
    const float* dpw     = (const float*)d_in[9];
    const float* dpb     = (const float*)d_in[10];
    const float* a_log   = (const float*)d_in[11];
    const float* Dpar    = (const float*)d_in[12];
    const float* opw     = (const float*)d_in[13];
    const float* nout_w  = (const float*)d_in[14];
    const float* nout_b  = (const float*)d_in[15];
    float* logits = (float*)d_out;

    // workspace carve-out
    char*  ws  = (char*)d_ws;
    size_t off = 0;
    auto carve = [&](size_t bytes) {
        void* p = ws + off;
        off += align_up(bytes);
        return p;
    };
    bf16_t* emb_bf  = (bf16_t*)carve((size_t)V_ * DM_ * 2);
    bf16_t* ipw_bf  = (bf16_t*)carve((size_t)NLAYER * 2 * DI_ * DM_ * 2);
    bf16_t* xpw_bf  = (bf16_t*)carve((size_t)NLAYER * 64 * DI_ * 2);
    bf16_t* dpw_bf  = (bf16_t*)carve((size_t)NLAYER * DI_ * DTR_ * 2);
    bf16_t* opw_bf  = (bf16_t*)carve((size_t)NLAYER * DM_ * DI_ * 2);
    float*  h       = (float*)carve((size_t)BL_ * DM_ * 4);
    bf16_t* x_bf    = (bf16_t*)carve((size_t)BL_ * DM_ * 2);
    float*  xz      = (float*)carve((size_t)BL_ * 2 * DI_ * 4);
    float*  xf      = (float*)carve((size_t)BL_ * DI_ * 4);
    bf16_t* xf_bf   = (bf16_t*)carve((size_t)BL_ * DI_ * 2);
    float*  dtbc    = (float*)carve((size_t)BL_ * 64 * 4);
    bf16_t* dtbc_bf = (bf16_t*)carve((size_t)BL_ * 64 * 2);
    float*  dtv     = (float*)carve((size_t)BL_ * DI_ * 4);
    bf16_t* y_bf    = (bf16_t*)carve((size_t)BL_ * DI_ * 2);
    bf16_t* hf_bf   = (bf16_t*)carve((size_t)BL_ * DM_ * 2);

    auto cvt = [&](const float* s, bf16_t* dst, int n) {
        k_f32_to_bf16<<<(n + 255) / 256, 256, 0, stream>>>(s, dst, n);
    };
    cvt(emb, emb_bf, V_ * DM_);
    cvt(ipw, ipw_bf, NLAYER * 2 * DI_ * DM_);
    cvt(xpw, xpw_bf, NLAYER * 64 * DI_);
    cvt(dpw, dpw_bf, NLAYER * DI_ * DTR_);
    cvt(opw, opw_bf, NLAYER * DM_ * DI_);

    // h = emb[ids] + pos
    {
        int n = BL_ * DM_;
        k_embed<<<(n + 255) / 256, 256, 0, stream>>>(ids, emb, pos, h);
    }

    for (int l = 0; l < NLAYER; l++) {
        const float*  nw    = norm_w + (size_t)l * DM_;
        const float*  nb    = norm_b + (size_t)l * DM_;
        const bf16_t* ipwl  = ipw_bf + (size_t)l * 2 * DI_ * DM_;
        const float*  cwl   = conv_w + (size_t)l * DI_ * DC_;
        const float*  cbl   = conv_b + (size_t)l * DI_;
        const bf16_t* xpwl  = xpw_bf + (size_t)l * 64 * DI_;
        const bf16_t* dpwl  = dpw_bf + (size_t)l * DI_ * DTR_;
        const float*  dpbl  = dpb + (size_t)l * DI_;
        const float*  alogl = a_log + (size_t)l * DI_ * DS_;
        const float*  Dl    = Dpar + (size_t)l * DI_;
        const bf16_t* opwl  = opw_bf + (size_t)l * DM_ * DI_;

        // 1) x = LN(h)  (bf16)
        k_layernorm<<<BL_, 128, 0, stream>>>(h, nw, nb, x_bf, DM_);

        // 2) xz = x @ ipw^T   [2048 x 2048], K=512
        {
            dim3 g(2 * DI_ / 64, BL_ / 64);
            k_gemm<false, false, 0, false><<<g, 128, 0, stream>>>(
                x_bf, ipwl, xz, nullptr, nullptr, nullptr, BL_, 2 * DI_, DM_, DM_);
        }

        // 3) depthwise conv + bias + SiLU -> xf (fp32 + bf16)
        {
            int n = BL_ * DI_;
            k_conv_silu<<<(n + 255) / 256, 256, 0, stream>>>(xz, cwl, cbl, xf, xf_bf);
        }

        // 4) dtbc = xf @ xpw^T   [2048 x 64], K=1024  (also bf16 copy)
        {
            dim3 g(64 / 64, BL_ / 64);
            k_gemm<false, false, 0, true><<<g, 128, 0, stream>>>(
                xf_bf, xpwl, dtbc, dtbc_bf, nullptr, nullptr, BL_, 64, DI_, DI_);
        }

        // 5) dt = softplus(dt_lo @ dpw^T + dpb)  [2048 x 1024], K=32, lda=64
        {
            dim3 g(DI_ / 64, BL_ / 64);
            k_gemm<true, false, 1, false><<<g, 128, 0, stream>>>(
                dtbc_bf, dpwl, dtv, nullptr, dpbl, nullptr, BL_, DI_, DTR_, 64);
        }

        // 6) selective scan + gate -> y (bf16)
        {
            int n = B_ * DI_;
            k_scan<<<(n + 255) / 256, 256, 0, stream>>>(dtv, xf, dtbc, xz, alogl,
                                                        Dl, y_bf);
        }

        // 7) h = y @ opw^T + h   [2048 x 512], K=1024
        {
            dim3 g(DM_ / 64, BL_ / 64);
            k_gemm<false, true, 0, false><<<g, 128, 0, stream>>>(
                y_bf, opwl, h, nullptr, nullptr, h, BL_, DM_, DI_, DI_);
        }
    }

    // final LN
    k_layernorm<<<BL_, 128, 0, stream>>>(h, nout_w, nout_b, hf_bf, DM_);

    // logits = hF @ emb^T   [2048 x 16384], K=512
    {
        dim3 g(V_ / 64, BL_ / 64);
        k_gemm<false, false, 0, false><<<g, 128, 0, stream>>>(
            hf_bf, emb_bf, logits, nullptr, nullptr, nullptr, BL_, V_, DM_, DM_);
    }
}
